// AdexLIF_68006512165484
// MI455X (gfx1250) — matrix operand order repaired and checked by reading, not compiled, not run on hardware
//
#include <hip/hip_runtime.h>
#include <cmath>

typedef __attribute__((ext_vector_type(16))) _Float16 v16h;
typedef __attribute__((ext_vector_type(8)))  _Float16 v8h;
typedef __attribute__((ext_vector_type(8)))  float    v8f;
typedef __attribute__((ext_vector_type(4)))  int      v4i;

// Explicit address-space pointer types: force global_load/global_store
// (round-1 asm showed everything lowering to flat_load_b128).
typedef __attribute__((address_space(1))) v4i      v4i_g;
typedef __attribute__((address_space(3))) v4i      v4i_l;
typedef __attribute__((address_space(1))) v8h      v8h_g;
typedef __attribute__((address_space(1))) float    float_g;

// ---- CDNA5 async global->LDS path (ASYNCcnt), guarded so both toolchains build ----
#if defined(__has_builtin)
#if __has_builtin(__builtin_amdgcn_global_load_async_to_lds_b128)
#define HAS_ASYNC_LDS 1
#endif
#if __has_builtin(__builtin_amdgcn_s_wait_asynccnt)
#define HAS_WAIT_ASYNC 1
#endif
#endif
#ifndef HAS_ASYNC_LDS
#define HAS_ASYNC_LDS 0
#endif
#ifndef HAS_WAIT_ASYNC
#define HAS_WAIT_ASYNC 0
#endif

#if HAS_ASYNC_LDS
#if HAS_WAIT_ASYNC
#define WAIT_ASYNC() __builtin_amdgcn_s_wait_asynccnt(0)
#else
#define WAIT_ASYNC() asm volatile("s_wait_asynccnt 0x0" ::: "memory")
#endif
#else
#define WAIT_ASYNC() ((void)0)
#endif

namespace {

constexpr int T_STEPS = 100;
constexpr int NBATCH  = 512;
constexpr int NK      = 512;
constexpr int NN      = 512;
constexpr int LDSW    = 520;   // padded f16 row stride (banks spread 4 apart)

constexpr float DT    = 1e-3f;
constexpr float I0    = 1e-3f;
constexpr float KAPPA = (0.75f + 0.66f) * 0.5f;
constexpr float UT    = 25.0e-3f;
constexpr float TAU_AMPA      = 2e-3f * UT / (KAPPA * 20.0f * I0);
constexpr float TAU_GABA_B    = 2e-3f * UT / (KAPPA * 5.0f * I0);
constexpr float TAU_SOMA      = 2e-3f * UT / (KAPPA * 5.0f * I0);
constexpr float TAU_AHP       = 4e-3f * UT / (KAPPA * 2.0f * I0);
constexpr float ISOMA_DPI_TAU = 5.0f * I0;
constexpr float SOMA_REFP     = 5.0f;
constexpr float ISOMA_RESET   = 1.2f * I0;
constexpr float VR            = ISOMA_DPI_TAU + I0;
constexpr float ISOMA_TH      = 2000.0f * I0;
constexpr float ISOMA_PFB_TH  = 1000.0f * I0;
constexpr float ALPHA_SOMA    = 4.0f;
constexpr float AMPA_GAIN     = 4.0f * 100.0f * I0;
constexpr float GABA_B_GAIN   = 4.0f * 100.0f * I0;
constexpr float AHP_GAIN      = 4.0f * 2.0f * I0;
constexpr float AHP_JUMP      = 4.0f * I0;
constexpr float ISOMA_PFB_NORM = 20.0f * I0;
constexpr float ISOMA_PFB_GAIN = 100.0f * I0;

union Frag { v16h v; v8h h[2]; };

} // namespace

// ---------------------------------------------------------------------------
// Prep: round weights to {0,1}, f32->f16, pre-swizzle into WMMA 32x16 (KxN)
// B-fragment layout: each lane's fragment = 32 contiguous bytes.
// ---------------------------------------------------------------------------
__global__ void __launch_bounds__(256)
prep_weights_kernel(const float* __restrict__ wA, const float* __restrict__ wG,
                    _Float16* __restrict__ outA, _Float16* __restrict__ outG)
{
    const int idx = blockIdx.x * 256 + threadIdx.x;   // 0 .. 512*512-1
    const int k = idx >> 9;
    const int n = idx & 511;
    const int kt = k >> 5, ko = k & 31;
    const int nt = n >> 4, nc = n & 15;
    const int lane = nc + ((ko & 8) ? 16 : 0);
    const int elem = (ko & 7) + ((ko & 16) >> 1);
    const int dst  = (((kt * 32 + nt) * 32 + lane) << 4) + elem;
    outA[dst] = (_Float16)rintf(wA[idx]);   // RNE == jnp.round on uniform(0,1)
    outG[dst] = (_Float16)rintf(wG[idx]);
}

// ---------------------------------------------------------------------------
// Fused AdexLIF: wave owns a 16x16 (batch x neuron) tile; state in registers
// for all 100 steps. Pipeline per step:
//   async(t+1 tile -> LDS f32 stage) overlaps [WMMA + state math of step t]
//   then s_wait_asynccnt + barrier + one cooperative f32->f16 convert pass
//   into double-buffered WMMA-ready f16 tiles.
// ---------------------------------------------------------------------------
__global__ void __launch_bounds__(256)
adexlif_step_kernel(const float* __restrict__ inA,   // [T,B,NK] f32
                    const float* __restrict__ inG,
                    const _Float16* __restrict__ wA, // pre-swizzled B fragments
                    const _Float16* __restrict__ wG,
                    float* __restrict__ Sout)        // [T,B,NN] f32
{
    __shared__ v4i      stageA[2048];           // raw 16x512 f32 tile (32 KB)
    __shared__ v4i      stageG[2048];
    __shared__ _Float16 ldsA16[2][16 * LDSW];   // double-buffered f16 tiles
    __shared__ _Float16 ldsG16[2][16 * LDSW];

    const int tid   = threadIdx.x;
    const int lane  = tid & 31;
    const int wave  = tid >> 5;
    const int b0    = blockIdx.x * 16;
    const int ntile = blockIdx.y * 8 + wave;    // neuron 16-col tile, 0..31

    // Per-lane state; C/D layout: VGPR v -> (M = v + 8*(lane>=16), N = lane%16)
    float mem[8], ampa[8], gaba[8], ahp[8], refr[8];
#pragma unroll
    for (int v = 0; v < 8; ++v) {
        mem[v] = 1.1f * I0; ampa[v] = I0; gaba[v] = I0; ahp[v] = 0.f; refr[v] = 0.f;
    }

    const int arow  = lane & 15;
    const int khalf = (lane < 16) ? 0 : 8;      // ISA 16-bit A-fragment K split
    const int nlo   = lane & 15;
    const int mhi   = (lane < 16) ? 0 : 8;

    // ---- fetch tile t into the f32 staging buffers (async if available) ----
    auto issue = [&](int t) {
#pragma unroll
        for (int i = 0; i < 8; ++i) {
            const int j = tid + i * 256;        // 16-byte chunk index in tile
            const float* sA = inA + ((size_t)t * NBATCH + b0) * NK + 4 * (size_t)j;
            const float* sG = inG + ((size_t)t * NBATCH + b0) * NK + 4 * (size_t)j;
#if HAS_ASYNC_LDS
            __builtin_amdgcn_global_load_async_to_lds_b128(
                (v4i_g*)sA, (v4i_l*)&stageA[j], 0, 0);
            __builtin_amdgcn_global_load_async_to_lds_b128(
                (v4i_g*)sG, (v4i_l*)&stageG[j], 0, 0);
#else
            stageA[j] = *(const v4i_g*)sA;      // sync fallback, still global_load
            stageG[j] = *(const v4i_g*)sG;
#endif
        }
    };

    // ---- cooperative f32 -> f16 convert into WMMA-ready padded tiles ----
    auto convert = [&](int buf) {
#pragma unroll
        for (int i = 0; i < 8; ++i) {
            const int j   = tid + i * 256;
            const int row = j >> 7;
            const int c4  = j & 127;
            union { v4i u; float f[4]; } ua, ug;
            ua.u = stageA[j];
            ug.u = stageG[j];
            union { _Float16 h[4]; uint2 u2; } ha, hg;
            ha.h[0] = (_Float16)ua.f[0]; ha.h[1] = (_Float16)ua.f[1];
            ha.h[2] = (_Float16)ua.f[2]; ha.h[3] = (_Float16)ua.f[3];
            hg.h[0] = (_Float16)ug.f[0]; hg.h[1] = (_Float16)ug.f[1];
            hg.h[2] = (_Float16)ug.f[2]; hg.h[3] = (_Float16)ug.f[3];
            *reinterpret_cast<uint2*>(&ldsA16[buf][row * LDSW + c4 * 4]) = ha.u2;
            *reinterpret_cast<uint2*>(&ldsG16[buf][row * LDSW + c4 * 4]) = hg.u2;
        }
    };

    // ---- prologue: tile 0 ready in LDS16[0], tile 1 in flight ----
    issue(0);
    WAIT_ASYNC();
    __syncthreads();
    convert(0);
    __syncthreads();
    if (T_STEPS > 1) issue(1);

    for (int t = 0; t < T_STEPS; ++t) {
        const int buf = t & 1;
        const _Float16* bA16 = &ldsA16[buf][0];
        const _Float16* bG16 = &ldsG16[buf][0];

        // Defeat LICM on the t-invariant weight-fragment loads (they must be
        // re-read each step from L2, not hoisted into 100s of live VGPRs).
        int koff = 0;
        asm volatile("" : "+s"(koff));

        // ---- K=512: 4 independent WMMA chains (A/G x loK/hiK) ----
        v8f accA0 = {0.f,0.f,0.f,0.f,0.f,0.f,0.f,0.f};
        v8f accG0 = {0.f,0.f,0.f,0.f,0.f,0.f,0.f,0.f};
        v8f accA1 = {0.f,0.f,0.f,0.f,0.f,0.f,0.f,0.f};
        v8f accG1 = {0.f,0.f,0.f,0.f,0.f,0.f,0.f,0.f};
#pragma unroll
        for (int kt = 0; kt < 8; ++kt) {
            const int k0 = kt, k1 = kt + 8;
            const _Float16* pA0 = bA16 + arow * LDSW + k0 * 32 + khalf;
            const _Float16* pG0 = bG16 + arow * LDSW + k0 * 32 + khalf;
            const _Float16* pA1 = bA16 + arow * LDSW + k1 * 32 + khalf;
            const _Float16* pG1 = bG16 + arow * LDSW + k1 * 32 + khalf;
            Frag aA0, aG0, aA1, aG1, bA0, bG0, bA1, bG1;
            aA0.h[0] = *reinterpret_cast<const v8h*>(pA0);
            aA0.h[1] = *reinterpret_cast<const v8h*>(pA0 + 16);
            aG0.h[0] = *reinterpret_cast<const v8h*>(pG0);
            aG0.h[1] = *reinterpret_cast<const v8h*>(pG0 + 16);
            aA1.h[0] = *reinterpret_cast<const v8h*>(pA1);
            aA1.h[1] = *reinterpret_cast<const v8h*>(pA1 + 16);
            aG1.h[0] = *reinterpret_cast<const v8h*>(pG1);
            aG1.h[1] = *reinterpret_cast<const v8h*>(pG1 + 16);
            const v8h_g* qA0 = (const v8h_g*)(wA + koff + ((((k0 * 32 + ntile) * 32) + lane) << 4));
            const v8h_g* qG0 = (const v8h_g*)(wG + koff + ((((k0 * 32 + ntile) * 32) + lane) << 4));
            const v8h_g* qA1 = (const v8h_g*)(wA + koff + ((((k1 * 32 + ntile) * 32) + lane) << 4));
            const v8h_g* qG1 = (const v8h_g*)(wG + koff + ((((k1 * 32 + ntile) * 32) + lane) << 4));
            bA0.h[0] = qA0[0]; bA0.h[1] = qA0[1];
            bG0.h[0] = qG0[0]; bG0.h[1] = qG0[1];
            bA1.h[0] = qA1[0]; bA1.h[1] = qA1[1];
            bG1.h[0] = qG1[0]; bG1.h[1] = qG1[1];
            accA0 = __builtin_amdgcn_wmma_f32_16x16x32_f16(
                        false, aA0.v, false, bA0.v, (short)0, accA0, false, false);
            accG0 = __builtin_amdgcn_wmma_f32_16x16x32_f16(
                        false, aG0.v, false, bG0.v, (short)0, accG0, false, false);
            accA1 = __builtin_amdgcn_wmma_f32_16x16x32_f16(
                        false, aA1.v, false, bA1.v, (short)0, accA1, false, false);
            accG1 = __builtin_amdgcn_wmma_f32_16x16x32_f16(
                        false, aG1.v, false, bG1.v, (short)0, accG1, false, false);
        }

        // ---- elementwise AdexLIF update + spike write (global_store_b32) ----
        float* srow = Sout + ((size_t)t * NBATCH + (b0 + mhi)) * NN + ntile * 16 + nlo;
#pragma unroll
        for (int v = 0; v < 8; ++v) {
            const float PA = accA0[v] + accA1[v];
            const float PG = accG0[v] + accG1[v];
            const float pfb   = ISOMA_PFB_GAIN /
                                (1.0f + __expf(-(mem[v] - ISOMA_PFB_TH) / ISOMA_PFB_NORM));
            const float ahp_d = ahp[v];
            const float dahp  = (-AHP_GAIN - ahp_d) /
                                (TAU_AHP * (1.0f + AHP_GAIN / ahp_d)); // ahp=0 -> -0 (ref)
            const float Iin   = fmaxf(ampa[v] - gaba[v] + I0, I0);
            const float Isum  = VR + ahp_d - pfb;
            const float dmem  = (ALPHA_SOMA * (Iin - Isum) - Isum * mem[v] / ISOMA_DPI_TAU) /
                                (TAU_SOMA * (1.0f + ALPHA_SOMA * ISOMA_DPI_TAU / mem[v]));
            const float dampa = (I0 - ampa[v]) / TAU_AMPA;      // pre-injection
            ampa[v] += AMPA_GAIN * PA;
            const float dgaba = (I0 - gaba[v]) / TAU_GABA_B;    // pre-injection
            gaba[v] += GABA_B_GAIN * PG;
            refr[v] -= (refr[v] > 0.f) ? 1.f : 0.f;
            mem[v]  += (refr[v] <= 0.f) ? DT * dmem : 0.f;
            ahp[v]   = ahp_d + DT * dahp;
            ampa[v] += DT * dampa;
            gaba[v] += DT * dgaba;
            const float s = (mem[v] > ISOMA_TH) ? 1.f : 0.f;
            refr[v] += floorf(s * SOMA_REFP);
            ahp[v]  += AHP_JUMP * s;
            mem[v]   = s * ISOMA_RESET + (1.f - s) * mem[v];
            mem[v]   = fmaxf(mem[v], I0);
            *(float_g*)(srow + (size_t)v * NN) = s;
        }

        // ---- rotate pipeline: finish async(t+1), convert it, launch t+2 ----
        if (t + 1 < T_STEPS) {
            WAIT_ASYNC();          // tile t+1 landed in stageA/stageG
            __syncthreads();       // all waves done reading LDS16[buf], async visible
            convert((t + 1) & 1);
            __syncthreads();       // converts visible; stage buffers free
            if (t + 2 < T_STEPS) issue(t + 2);
        }
    }
}

// ---------------------------------------------------------------------------
extern "C" void kernel_launch(void* const* d_in, const int* in_sizes, int n_in,
                              void* d_out, int out_size, void* d_ws, size_t ws_size,
                              hipStream_t stream) {
    (void)in_sizes; (void)n_in; (void)out_size; (void)ws_size;
    const float* inA = (const float*)d_in[0];   // input_ampa   [T,B,N_IN]
    const float* inG = (const float*)d_in[1];   // input_gaba_b [T,B,N_IN]
    const float* wA  = (const float*)d_in[2];   // weight_ampa  [N_IN,N_OUT]
    const float* wG  = (const float*)d_in[3];   // weight_gaba_b

    _Float16* wsA = (_Float16*)d_ws;            // 0.5 MB
    _Float16* wsG = wsA + (size_t)NK * NN;      // 0.5 MB

    prep_weights_kernel<<<dim3((NK * NN) / 256), dim3(256), 0, stream>>>(wA, wG, wsA, wsG);

    adexlif_step_kernel<<<dim3(NBATCH / 16, NN / 128), dim3(256), 0, stream>>>(
        inA, inG, wsA, wsG, (float*)d_out);
}
